// PhysicalMatformer_22488448762661
// MI455X (gfx1250) — matrix operand order repaired and checked by reading, not compiled, not run on hardware
//
#include <hip/hip_runtime.h>
#include <hip/hip_bf16.h>
#include <math.h>

typedef _Float16 half_t;
typedef __attribute__((ext_vector_type(16))) _Float16 v16h;
typedef __attribute__((ext_vector_type(8)))  _Float16 v8h;
typedef __attribute__((ext_vector_type(8)))  float    v8f;

#define NNODES 8192
#define NEDGES 65536
#define NGRAPH 256

#if __has_builtin(__builtin_amdgcn_global_load_async_to_lds_b128)
#define HAVE_ASYNC_LDS 1
#else
#define HAVE_ASYNC_LDS 0
#endif

typedef int v4i_t __attribute__((vector_size(16)));
typedef __attribute__((address_space(1))) v4i_t* gv4i_p;
typedef __attribute__((address_space(3))) v4i_t* lv4i_p;

// copy 16B global -> LDS (async path on gfx1250, sync fallback)
__device__ __forceinline__ void stage16(const half_t* g, half_t* l)
{
#if HAVE_ASYNC_LDS
  __builtin_amdgcn_global_load_async_to_lds_b128(
      (gv4i_p)const_cast<half_t*>(g), (lv4i_p)l, 0, 0);
#else
  *(v8h*)l = *(const v8h*)g;
#endif
}

template<int PEND>
__device__ __forceinline__ void wait_async()
{
#if HAVE_ASYNC_LDS
#if __has_builtin(__builtin_amdgcn_s_wait_asynccnt)
  __builtin_amdgcn_s_wait_asynccnt(PEND);
#else
  asm volatile("s_wait_asynccnt %0" :: "i"(PEND));
#endif
#endif
}

// ---------------------------------------------------------------------------
// WMMA GEMM:  C[M,N] = act(A[M,Kp](f16) @ Bt[Npad,Kp]^T(f16) + bias) (*gate)
// Block = 128 threads = 4 waves. Each wave: one 16-row M tile x 64 N columns
// (4 x v8f accumulators, A fragment reused across 4 WMMAs). The 64-column B
// strip is shared by all 4 waves and double-buffered in LDS via async copies.
// VGPR layouts used here (A/B k contiguous per lane half, not the ISA table order):
//   A 16x32 f16 : lanes 0-15 row M=l, K = kk+0..15 ; lanes 16-31 K=kk+16..31
//   B 32x16 f16 : lanes 0-15 col N=l, K = kk+0..15 ; lanes 16-31 K=kk+16..31
//   C 16x16 f32 : vgpr r -> M = 8*(lane>>4)+r, N = lane&15
// Requires: M % 64 == 0, Kp % 32 == 0, Bt has ceil64(N) rows.
// ---------------------------------------------------------------------------
template<int ACT, bool GATE>
__global__ __launch_bounds__(128)
void k_gemm(const half_t* __restrict__ A, const half_t* __restrict__ Bt,
            const float* __restrict__ bias, const half_t* __restrict__ gate,
            float* __restrict__ Cf, half_t* __restrict__ Ch,
            int M, int N, int Kp, int ldc)
{
  (void)M;
  const int lane   = threadIdx.x & 31;
  const int wave   = threadIdx.x >> 5;
  const int half16 = lane >> 4;
  const int l15    = lane & 15;
  const int akg    = half16 * 16;   // A k sub-offset
  const int bkg    = half16 * 16;   // B k sub-offset

  const int mTile = blockIdx.y * 4 + wave;
  const int nBase = blockIdx.x * 64;

  const half_t* Arow   = A  + (size_t)(mTile * 16 + l15) * Kp;
  const half_t* Bstrip = Bt + (size_t)nBase * Kp;

  // 64 rows x 32 halfs, padded to 40-half pitch (20 dword banks -> conflict free)
  __shared__ __align__(16) half_t Bs[2][64 * 40];

  auto stageB = [&](int buf, int kk) {
#pragma unroll
    for (int u = 0; u < 2; ++u) {
      int c = threadIdx.x + u * 128;          // 256 chunks of 16B
      int row = c >> 2, part = c & 3;
      stage16(Bstrip + (size_t)row * Kp + kk + part * 8,
              &Bs[buf][row * 40 + part * 8]);
    }
  };

  v8f acc[4] = {{}, {}, {}, {}};
  const int steps = Kp >> 5;

  stageB(0, 0);
  for (int i = 0; i < steps; ++i) {
    const int cur = i & 1;
    if (i + 1 < steps) { stageB(cur ^ 1, (i + 1) << 5); wait_async<2>(); }
    else               { wait_async<0>(); }
    __syncthreads();                           // Bs[cur] ready for everyone

    const int kk = i << 5;
    v8h a0 = *(const v8h*)(Arow + kk + akg);
    v8h a1 = *(const v8h*)(Arow + kk + akg + 8);
    if (i + 2 < steps) __builtin_prefetch(Arow + kk + 64 + akg, 0, 0);
    v16h av;
#pragma unroll
    for (int j = 0; j < 8; ++j) { av[j] = a0[j]; av[j + 8] = a1[j]; }

#pragma unroll
    for (int nt = 0; nt < 4; ++nt) {
      const half_t* bp = &Bs[cur][(nt * 16 + l15) * 40 + bkg];
      v8h b0 = *(const v8h*)bp;
      v8h b1 = *(const v8h*)(bp + 8);
      v16h bv;
#pragma unroll
      for (int j = 0; j < 8; ++j) { bv[j] = b0[j]; bv[j + 8] = b1[j]; }
      acc[nt] = __builtin_amdgcn_wmma_f32_16x16x32_f16(false, av, false, bv,
                                                       (short)0, acc[nt], false, false);
    }
    __syncthreads();                           // reads done before overwrite
  }

#pragma unroll
  for (int nt = 0; nt < 4; ++nt) {
    const int nOut = nBase + nt * 16 + l15;
    if (nOut < N) {
      const float bs = bias ? bias[nOut] : 0.f;
#pragma unroll
      for (int r = 0; r < 8; ++r) {
        const int mOut = mTile * 16 + half16 * 8 + r;
        float v = acc[nt][r] + bs;
        if (GATE) v *= (float)gate[(size_t)mOut * ldc + nOut];
        if (ACT == 1) v = (v > 20.f) ? v : log1pf(__expf(v));   // softplus
        else if (ACT == 2) v = v / (1.f + __expf(-v));          // silu
        const size_t idx = (size_t)mOut * ldc + nOut;
        if (Cf) Cf[idx] = v;
        if (Ch) Ch[idx] = (half_t)v;
      }
    }
  }
}

// f32 [M,K] -> f16 [M,Kp] zero-padded
__global__ void k_cvt_pad(const float* __restrict__ A, half_t* __restrict__ B,
                          int M, int K, int Kp)
{
  int i = blockIdx.x * 256 + threadIdx.x;
  if (i >= M * Kp) return;
  int r = i / Kp, c = i - r * Kp;
  B[i] = (c < K) ? (half_t)A[(size_t)r * K + c] : (half_t)0.f;
}

// W f32 [K,N] -> Bt f16 [Np,Kp]  (Bt[n*Kp+k] = W[k*N+n], zero-padded)
__global__ void k_wt(const float* __restrict__ W, half_t* __restrict__ Bt,
                     int K, int N, int Kp, int Np)
{
  int i = blockIdx.x * 256 + threadIdx.x;
  if (i >= Np * Kp) return;
  int n = i / Kp, k = i - n * Kp;
  Bt[i] = (n < N && k < K) ? (half_t)W[(size_t)k * N + n] : (half_t)0.f;
}

__global__ void k_zero(float* p, int n)
{
  int i = blockIdx.x * 256 + threadIdx.x;
  if (i < n) p[i] = 0.f;
}

// RBF(|edge_attr|, 0..8, 128 bins) -> f16 [E,128]
__global__ void k_rbf_edge(const float* __restrict__ ea, half_t* __restrict__ out)
{
  int e = blockIdx.x, j = threadIdx.x;            // 128 threads
  float x = ea[e * 3], y = ea[e * 3 + 1], z = ea[e * 3 + 2];
  float d = sqrtf(x * x + y * y + z * z);
  float c = 8.f * (float)j / 127.f;
  float t = d - c;
  out[(size_t)e * 128 + j] = (half_t)__expf(-(127.f / 8.f) * t * t);
}

// RBF over 768 lattice row-vector norms -> f16 [768,128]
__global__ void k_rbf_lat(const float* __restrict__ lat, half_t* __restrict__ out)
{
  int r = blockIdx.x, j = threadIdx.x;
  float x = lat[r * 3], y = lat[r * 3 + 1], z = lat[r * 3 + 2];
  float d = sqrtf(x * x + y * y + z * z);
  float c = 8.f * (float)j / 127.f;
  float t = d - c;
  out[(size_t)r * 128 + j] = (half_t)__expf(-(127.f / 8.f) * t * t);
}

// Angle RBFs: cos(gamma/beta/alpha) -> f16 [256,64] (40 bins, padded to 64)
__global__ void k_rbf_ang(const float* __restrict__ lat,
                          half_t* og, half_t* ob, half_t* oa)
{
  int g = blockIdx.x, j = threadIdx.x;            // 64 threads
  const float* L = lat + g * 9;
  float n1 = sqrtf(L[0]*L[0] + L[1]*L[1] + L[2]*L[2]);
  float n2 = sqrtf(L[3]*L[3] + L[4]*L[4] + L[5]*L[5]);
  float n3 = sqrtf(L[6]*L[6] + L[7]*L[7] + L[8]*L[8]);
  float d12 = L[0]*L[3] + L[1]*L[4] + L[2]*L[5];
  float d13 = L[0]*L[6] + L[1]*L[7] + L[2]*L[8];
  float d23 = L[3]*L[6] + L[4]*L[7] + L[5]*L[8];
  float cg = fminf(1.f, fmaxf(-1.f, d12 / (n1 * n2)));
  float cb = fminf(1.f, fmaxf(-1.f, d13 / (n1 * n3)));
  float ca = fminf(1.f, fmaxf(-1.f, d23 / (n2 * n3)));
  float center = -1.f + (float)j * (2.f / 39.f);
  const float gam = 19.5f;
  float tg = cg - center, tb = cb - center, ta = ca - center;
  bool ok = (j < 40);
  og[(size_t)g * 64 + j] = ok ? (half_t)__expf(-gam * tg * tg) : (half_t)0.f;
  ob[(size_t)g * 64 + j] = ok ? (half_t)__expf(-gam * tb * tb) : (half_t)0.f;
  oa[(size_t)g * 64 + j] = ok ? (half_t)__expf(-gam * ta * ta) : (half_t)0.f;
}

// lat_edge [768,128] -> latcat [256,768] columns 0..383
__global__ void k_pack_latedge(const half_t* __restrict__ s, half_t* __restrict__ d)
{
  int i = blockIdx.x * 256 + threadIdx.x;
  if (i >= 256 * 384) return;
  int g = i / 384, c = i - g * 384;
  d[(size_t)g * 768 + c] = s[i];
}

// lat_emb[batch[i]] -> nodecat columns 128..255
__global__ void k_gather_lat(const half_t* __restrict__ le, const int* __restrict__ batch,
                             half_t* __restrict__ nodecat)
{
  int i = blockIdx.x, c = threadIdx.x;           // 128 threads
  nodecat[(size_t)i * 256 + 128 + c] = le[(size_t)batch[i] * 128 + c];
}

// Fused per-(edge,head) attention: alpha = q3*key_cat/sqrt(384); gate = sigmoid(LN(alpha))
__global__ __launch_bounds__(128)
void k_edge_att(const half_t* __restrict__ q, const half_t* __restrict__ k,
                const half_t* __restrict__ v, const half_t* __restrict__ e,
                const int* __restrict__ src, const int* __restrict__ dst,
                const float* __restrict__ lng, const float* __restrict__ lnb,
                half_t* __restrict__ valcat, half_t* __restrict__ gate)
{
  const int edge = blockIdx.x;
  const int h    = threadIdx.x >> 5;
  const int lane = threadIdx.x & 31;
  const int s = src[edge], d = dst[edge];
  const half_t* qd = q + (size_t)d * 512 + h * 128;
  const half_t* kd = k + (size_t)d * 512 + h * 128;
  const half_t* ks = k + (size_t)s * 512 + h * 128;
  const half_t* vd = v + (size_t)d * 512 + h * 128;
  const half_t* vs = v + (size_t)s * 512 + h * 128;
  const half_t* ee = e + (size_t)edge * 512 + h * 128;

  float al[12], vc[12];
  const float scale = rsqrtf(384.f);
  float sum = 0.f, sq = 0.f;
#pragma unroll
  for (int t = 0; t < 12; ++t) {
    int j = lane * 12 + t;                       // 0..383
    int part = j >> 7, c = j & 127;
    float kc = (part == 0) ? (float)kd[c] : (part == 1) ? (float)ks[c] : (float)ee[c];
    float vv = (part == 0) ? (float)vd[c] : (part == 1) ? (float)vs[c] : (float)ee[c];
    float a = (float)qd[c] * kc * scale;
    al[t] = a; vc[t] = vv;
    sum += a; sq += a * a;
  }
#pragma unroll
  for (int o = 16; o > 0; o >>= 1) {
    sum += __shfl_xor(sum, o, 32);
    sq  += __shfl_xor(sq,  o, 32);
  }
  const float mean = sum / 384.f;
  const float inv  = rsqrtf(sq / 384.f - mean * mean + 1e-5f);
  const size_t row = ((size_t)edge * 4 + h) * 384;
#pragma unroll
  for (int t = 0; t < 12; ++t) {
    int j = lane * 12 + t;
    float ln = (al[t] - mean) * inv * lng[j] + lnb[j];
    gate[row + j]   = (half_t)(1.f / (1.f + __expf(-ln)));
    valcat[row + j] = (half_t)vc[t];
  }
}

// LayerNorm(msg row of 128) then scatter-add into agg[dst, h*128+c]
__global__ __launch_bounds__(128)
void k_msgln_scatter(const float* __restrict__ msg, const float* __restrict__ g,
                     const float* __restrict__ b, const int* __restrict__ dst,
                     float* __restrict__ agg)
{
  const int row = blockIdx.x;                    // edge*4 + h
  const int c = threadIdx.x;
  float x = msg[(size_t)row * 128 + c];
  __shared__ float s1[128], s2[128];
  s1[c] = x; s2[c] = x * x;
  __syncthreads();
  for (int st = 64; st > 0; st >>= 1) {
    if (c < st) { s1[c] += s1[c + st]; s2[c] += s2[c + st]; }
    __syncthreads();
  }
  float mean = s1[0] / 128.f;
  float inv  = rsqrtf(s2[0] / 128.f - mean * mean + 1e-5f);
  float y = (x - mean) * inv * g[c] + b[c];
  int edge = row >> 2, h = row & 3;
  atomicAdd(&agg[(size_t)dst[edge] * 512 + h * 128 + c], y);
}

// BatchNorm stats over 8192 rows, per channel
__global__ __launch_bounds__(256)
void k_bn_stats(const float* __restrict__ x, float* __restrict__ stats)
{
  const int ch = blockIdx.x;                     // 128 channels
  float s = 0.f, s2 = 0.f;
  for (int r = threadIdx.x; r < NNODES; r += 256) {
    float v = x[(size_t)r * 128 + ch];
    s += v; s2 += v * v;
  }
  __shared__ float a[256], b[256];
  a[threadIdx.x] = s; b[threadIdx.x] = s2;
  __syncthreads();
  for (int st = 128; st > 0; st >>= 1) {
    if (threadIdx.x < st) { a[threadIdx.x] += a[threadIdx.x + st]; b[threadIdx.x] += b[threadIdx.x + st]; }
    __syncthreads();
  }
  if (threadIdx.x == 0) {
    float m = a[0] / (float)NNODES;
    stats[ch] = m;
    stats[128 + ch] = b[0] / (float)NNODES - m * m;
  }
}

__global__ void k_bn_silu(const float* __restrict__ x, const float* __restrict__ stats,
                          const float* __restrict__ g, const float* __restrict__ b,
                          half_t* __restrict__ out)
{
  int i = blockIdx.x * 256 + threadIdx.x;
  if (i >= NNODES * 128) return;
  int ch = i & 127;
  float y = (x[i] - stats[ch]) * rsqrtf(stats[128 + ch] + 1e-5f) * g[ch] + b[ch];
  out[i] = (half_t)(y / (1.f + __expf(-y)));
}

__global__ void k_count(const int* __restrict__ batch, float* __restrict__ counts)
{
  int i = blockIdx.x * 256 + threadIdx.x;
  if (i < NNODES) atomicAdd(&counts[batch[i]], 1.f);
}

__global__ void k_pool(const half_t* __restrict__ node, const int* __restrict__ batch,
                       float* __restrict__ feats)
{
  int i = blockIdx.x, c = threadIdx.x;           // 128 threads
  atomicAdd(&feats[(size_t)batch[i] * 128 + c], (float)node[(size_t)i * 128 + c]);
}

__global__ void k_finalize_feats(const float* __restrict__ feats, const float* __restrict__ counts,
                                 const float* __restrict__ latemb, half_t* __restrict__ out)
{
  int i = blockIdx.x * 256 + threadIdx.x;
  if (i >= NGRAPH * 128) return;
  int g = i >> 7;
  out[i] = (half_t)(feats[i] / fmaxf(counts[g], 1.f) + latemb[i]);
}

// ---------------------------------------------------------------------------
extern "C" void kernel_launch(void* const* d_in, const int* in_sizes, int n_in,
                              void* d_out, int out_size, void* d_ws, size_t ws_size,
                              hipStream_t stream)
{
  (void)in_sizes; (void)n_in; (void)out_size; (void)ws_size;
  auto F = [&](int i) { return (const float*)d_in[i]; };
  const int* edge_index = (const int*)d_in[129];
  const int* batch      = (const int*)d_in[130];
  const int* esrc = edge_index;
  const int* edst = edge_index + NEDGES;

  char* base = (char*)d_ws;
  size_t off = 0;
  auto alloc = [&](size_t elems, size_t esz) -> void* {
    off = (off + 255) & ~(size_t)255;
    void* p = base + off;
    off += elems * esz;
    return p;
  };

  // ---- activation buffers -------------------------------------------------
  half_t* x_pad   = (half_t*)alloc((size_t)NNODES * 96, 2);
  half_t* rbf_e   = (half_t*)alloc((size_t)NEDGES * 128, 2);
  half_t* tmp_e   = (half_t*)alloc((size_t)NEDGES * 128, 2);
  half_t* edgeft  = (half_t*)alloc((size_t)NEDGES * 128, 2);
  half_t* latv    = (half_t*)alloc(768 * 128, 2);
  half_t* latv2   = (half_t*)alloc(768 * 128, 2);
  half_t* latedge = (half_t*)alloc(768 * 128, 2);
  half_t* angg    = (half_t*)alloc(NGRAPH * 64, 2);
  half_t* angb    = (half_t*)alloc(NGRAPH * 64, 2);
  half_t* anga    = (half_t*)alloc(NGRAPH * 64, 2);
  half_t* angtmp  = (half_t*)alloc(NGRAPH * 128, 2);
  half_t* latcat  = (half_t*)alloc(NGRAPH * 768, 2);
  half_t* lattmp  = (half_t*)alloc(NGRAPH * 128, 2);
  half_t* latemb_h= (half_t*)alloc(NGRAPH * 128, 2);
  float*  latemb_f= (float*) alloc(NGRAPH * 128, 4);
  half_t* nodecat = (half_t*)alloc((size_t)NNODES * 256, 2);
  half_t* nodetmp = (half_t*)alloc((size_t)NNODES * 128, 2);
  half_t* node_h  = (half_t*)alloc((size_t)NNODES * 128, 2);
  half_t* q_h     = (half_t*)alloc((size_t)NNODES * 512, 2);
  half_t* k_h     = (half_t*)alloc((size_t)NNODES * 512, 2);
  half_t* v_h     = (half_t*)alloc((size_t)NNODES * 512, 2);
  half_t* e_h     = (half_t*)alloc((size_t)NEDGES * 512, 2);
  half_t* valcat  = (half_t*)alloc((size_t)NEDGES * 4 * 384, 2);
  half_t* gate    = (half_t*)alloc((size_t)NEDGES * 4 * 384, 2);
  half_t* mmsg    = (half_t*)alloc((size_t)NEDGES * 4 * 384, 2);
  float*  msg_f   = (float*) alloc((size_t)NEDGES * 4 * 128, 4);
  float*  agg_f   = (float*) alloc((size_t)NNODES * 512, 4);
  half_t* agg_h   = (half_t*)alloc((size_t)NNODES * 512, 2);
  float*  convout = (float*) alloc((size_t)NNODES * 128, 4);
  float*  bnstats = (float*) alloc(256, 4);
  float*  feats_f = (float*) alloc(NGRAPH * 128, 4);
  float*  counts  = (float*) alloc(NGRAPH, 4);
  half_t* feats_h = (half_t*)alloc(NGRAPH * 128, 2);
  half_t* feats2_h= (half_t*)alloc(NGRAPH * 128, 2);

  // ---- transposed f16 weights --------------------------------------------
  half_t* atom_t    = (half_t*)alloc(128 * 96, 2);
  half_t* rbf1_t    = (half_t*)alloc(128 * 128, 2);
  half_t* rbf2_t    = (half_t*)alloc(128 * 128, 2);
  half_t* latrbf1_t = (half_t*)alloc(128 * 128, 2);
  half_t* latrbf2_t = (half_t*)alloc(128 * 128, 2);
  half_t* latang1_t = (half_t*)alloc(128 * 64, 2);
  half_t* latang2_t = (half_t*)alloc(128 * 128, 2);
  half_t* latemb1_t = (half_t*)alloc(128 * 768, 2);
  half_t* latemb2_t = (half_t*)alloc(128 * 128, 2);
  half_t* latat1_t  = (half_t*)alloc(128 * 256, 2);
  half_t* latat2_t  = (half_t*)alloc(128 * 128, 2);
  half_t* fc_t      = (half_t*)alloc(128 * 128, 2);
  half_t* head_t    = (half_t*)alloc(256 * 128, 2);
  half_t *wk_t[5], *wq_t[5], *wv_t[5], *we_t[5], *wc_t[5], *wmu_t[5], *wml_t[5];
  for (int l = 0; l < 5; ++l) {
    wk_t[l]  = (half_t*)alloc(512 * 128, 2);
    wq_t[l]  = (half_t*)alloc(512 * 128, 2);
    wv_t[l]  = (half_t*)alloc(512 * 128, 2);
    we_t[l]  = (half_t*)alloc(512 * 128, 2);
    wc_t[l]  = (half_t*)alloc(128 * 512, 2);
    wmu_t[l] = (half_t*)alloc(384 * 384, 2);
    wml_t[l] = (half_t*)alloc(128 * 384, 2);
  }

  auto wt = [&](const float* W, half_t* Bt, int K, int N, int Kp, int Np) {
    int n = Np * Kp;
    k_wt<<<(n + 255) / 256, 256, 0, stream>>>(W, Bt, K, N, Kp, Np);
  };
  // GEMM launcher: act 0=none 1=softplus 2=silu
  auto gemm = [&](const half_t* A, const half_t* Bt, const float* bias,
                  const half_t* g, float* Cf, half_t* Ch,
                  int M, int N, int Kp, int ldc, int act) {
    dim3 grid((N + 63) >> 6, M >> 6);
    if (g)             k_gemm<0, true ><<<grid, 128, 0, stream>>>(A, Bt, bias, g, Cf, Ch, M, N, Kp, ldc);
    else if (act == 1) k_gemm<1, false><<<grid, 128, 0, stream>>>(A, Bt, bias, g, Cf, Ch, M, N, Kp, ldc);
    else if (act == 2) k_gemm<2, false><<<grid, 128, 0, stream>>>(A, Bt, bias, g, Cf, Ch, M, N, Kp, ldc);
    else               k_gemm<0, false><<<grid, 128, 0, stream>>>(A, Bt, bias, g, Cf, Ch, M, N, Kp, ldc);
  };

  // ---- weight prep --------------------------------------------------------
  wt(F(3),  atom_t,    92, 128, 96, 128);
  wt(F(5),  rbf1_t,   128, 128, 128, 128);
  wt(F(7),  rbf2_t,   128, 128, 128, 128);
  wt(F(9),  latrbf1_t,128, 128, 128, 128);
  wt(F(11), latrbf2_t,128, 128, 128, 128);
  wt(F(13), latang1_t, 40, 128, 64, 128);
  wt(F(15), latang2_t,128, 128, 128, 128);
  wt(F(17), latemb1_t,768, 128, 768, 128);
  wt(F(19), latemb2_t,128, 128, 128, 128);
  wt(F(21), latat1_t, 256, 128, 256, 128);
  wt(F(23), latat2_t, 128, 128, 128, 128);
  for (int l = 0; l < 5; ++l) {
    int cb = 25 + l * 20;
    wt(F(cb + 0),  wk_t[l],  128, 512, 128, 512);
    wt(F(cb + 2),  wq_t[l],  128, 512, 128, 512);
    wt(F(cb + 4),  wv_t[l],  128, 512, 128, 512);
    wt(F(cb + 6),  we_t[l],  128, 512, 128, 512);
    wt(F(cb + 8),  wc_t[l],  512, 128, 512, 128);
    wt(F(cb + 10), wmu_t[l], 384, 384, 384, 384);
    wt(F(cb + 12), wml_t[l], 384, 128, 384, 128);
  }
  wt(F(125), fc_t,   128, 128, 128, 128);
  wt(F(127), head_t, 128, 200, 128, 256);

  // ---- embeddings ---------------------------------------------------------
  k_cvt_pad<<<(NNODES * 96 + 255) / 256, 256, 0, stream>>>(F(0), x_pad, NNODES, 92, 96);

  // edge features: rbf -> softplus(lin) -> lin
  k_rbf_edge<<<NEDGES, 128, 0, stream>>>(F(1), rbf_e);
  gemm(rbf_e, rbf1_t, F(6), nullptr, nullptr, tmp_e, NEDGES, 128, 128, 128, 1);
  gemm(tmp_e, rbf2_t, F(8), nullptr, nullptr, edgeft, NEDGES, 128, 128, 128, 0);

  // lattice edge features (768 rows) -> latcat[:, :384]
  k_rbf_lat<<<768, 128, 0, stream>>>(F(2), latv);
  gemm(latv,  latrbf1_t, F(10), nullptr, nullptr, latv2,   768, 128, 128, 128, 1);
  gemm(latv2, latrbf2_t, F(12), nullptr, nullptr, latedge, 768, 128, 128, 128, 0);
  k_pack_latedge<<<(256 * 384 + 255) / 256, 256, 0, stream>>>(latedge, latcat);

  // angle embeddings -> latcat[:, 384+t*128]
  k_rbf_ang<<<NGRAPH, 64, 0, stream>>>(F(2), angg, angb, anga);
  const half_t* angs[3] = { angg, angb, anga };
  for (int t = 0; t < 3; ++t) {
    gemm(angs[t], latang1_t, F(14), nullptr, nullptr, angtmp, NGRAPH, 128, 64, 128, 1);
    gemm(angtmp,  latang2_t, F(16), nullptr, nullptr, latcat + 384 + t * 128,
         NGRAPH, 128, 128, 768, 0);
  }

  // lattice embedding
  gemm(latcat, latemb1_t, F(18), nullptr, nullptr, lattmp, NGRAPH, 128, 768, 128, 1);
  gemm(lattmp, latemb2_t, F(20), nullptr, latemb_f, latemb_h, NGRAPH, 128, 128, 128, 0);

  // node = mlp2(lat_atom, [atom_emb(x) | lat_emb[batch]])
  gemm(x_pad, atom_t, F(4), nullptr, nullptr, nodecat, NNODES, 128, 96, 256, 0);
  k_gather_lat<<<NNODES, 128, 0, stream>>>(latemb_h, batch, nodecat);
  gemm(nodecat, latat1_t, F(22), nullptr, nullptr, nodetmp, NNODES, 128, 256, 128, 1);
  gemm(nodetmp, latat2_t, F(24), nullptr, nullptr, node_h,  NNODES, 128, 128, 128, 0);

  // ---- 5 matformer conv layers -------------------------------------------
  for (int l = 0; l < 5; ++l) {
    int cb = 25 + l * 20;
    gemm(node_h, wq_t[l], F(cb + 3), nullptr, nullptr, q_h, NNODES, 512, 128, 512, 0);
    gemm(node_h, wk_t[l], F(cb + 1), nullptr, nullptr, k_h, NNODES, 512, 128, 512, 0);
    gemm(node_h, wv_t[l], F(cb + 5), nullptr, nullptr, v_h, NNODES, 512, 128, 512, 0);
    gemm(edgeft, we_t[l], F(cb + 7), nullptr, nullptr, e_h, NEDGES, 512, 128, 512, 0);

    k_edge_att<<<NEDGES, 128, 0, stream>>>(q_h, k_h, v_h, e_h, esrc, edst,
                                           F(cb + 14), F(cb + 15), valcat, gate);

    // m = (val_cat @ Wmu + b) * gate ; msg = m @ Wml + b
    gemm(valcat, wmu_t[l], F(cb + 11), gate,    nullptr, mmsg, NEDGES * 4, 384, 384, 384, 0);
    gemm(mmsg,   wml_t[l], F(cb + 13), nullptr, msg_f,   nullptr, NEDGES * 4, 128, 384, 128, 0);

    k_zero<<<(NNODES * 512 + 255) / 256, 256, 0, stream>>>(agg_f, NNODES * 512);
    k_msgln_scatter<<<NEDGES * 4, 128, 0, stream>>>(msg_f, F(cb + 16), F(cb + 17), edst, agg_f);

    k_cvt_pad<<<(NNODES * 512 + 255) / 256, 256, 0, stream>>>(agg_f, agg_h, NNODES, 512, 512);
    gemm(agg_h, wc_t[l], F(cb + 9), nullptr, convout, nullptr, NNODES, 128, 512, 128, 0);

    k_bn_stats<<<128, 256, 0, stream>>>(convout, bnstats);
    k_bn_silu<<<(NNODES * 128 + 255) / 256, 256, 0, stream>>>(convout, bnstats,
                                                              F(cb + 18), F(cb + 19), node_h);
  }

  // ---- readout ------------------------------------------------------------
  k_zero<<<(NGRAPH * 128 + 255) / 256, 256, 0, stream>>>(feats_f, NGRAPH * 128);
  k_zero<<<1, 256, 0, stream>>>(counts, NGRAPH);
  k_count<<<(NNODES + 255) / 256, 256, 0, stream>>>(batch, counts);
  k_pool<<<NNODES, 128, 0, stream>>>(node_h, batch, feats_f);
  k_finalize_feats<<<(NGRAPH * 128 + 255) / 256, 256, 0, stream>>>(feats_f, counts,
                                                                   latemb_f, feats_h);
  gemm(feats_h, fc_t, F(126), nullptr, nullptr, feats2_h, NGRAPH, 128, 128, 128, 2);
  gemm(feats2_h, head_t, F(128), nullptr, (float*)d_out, nullptr, NGRAPH, 200, 128, 200, 0);
}
